// AULoss_37151467110996
// MI455X (gfx1250) — matrix operand order repaired and checked by reading, not compiled, not run on hardware
//
#include <hip/hip_runtime.h>
#include <hip/hip_bf16.h>
#include <math.h>

typedef __attribute__((ext_vector_type(16))) __bf16 v16bf;
typedef __attribute__((ext_vector_type(8)))  float  v8f;

constexpr int kN    = 8192;
constexpr int kD    = 512;
constexpr int kBT   = 128;                      // I/J tile size
constexpr int kKC   = 32;                       // K chunk (one WMMA K)
constexpr int kNCH  = kD / kKC;                 // 16 chunks
constexpr int kNBLK = kN / kBT;                 // 64
constexpr int kNTRI = kNBLK * (kNBLK + 1) / 2;  // 2080 upper-tri tiles
constexpr int kST   = 40;                       // LDS row stride (bf16 elems), 80B rows
constexpr int kASZ  = kBT * kST * 2;            // bytes per LDS tile array (10240)

// ---- bf16 helpers (round-to-nearest-even), no reliance on __bf16 arithmetic ----
__device__ __forceinline__ unsigned short f32_bf16(float x) {
    unsigned u = __float_as_uint(x);
    u += 0x7fffu + ((u >> 16) & 1u);
    return (unsigned short)(u >> 16);
}
__device__ __forceinline__ float bf16_f32(unsigned short s) {
    return __uint_as_float(((unsigned)s) << 16);
}

union Frag {
    v16bf v;
    uint4 q[2];
};

// ---- CDNA5 async global->LDS copy (bypasses VGPRs, tracked by ASYNCcnt) ----
__device__ __forceinline__ void async_copy_b128(unsigned lds_off, const void* gaddr) {
    asm volatile("global_load_async_to_lds_b128 %0, %1, off"
                 :: "v"(lds_off), "v"(gaddr)
                 : "memory");
}
__device__ __forceinline__ void wait_async_all() {
    asm volatile("s_wait_asynccnt 0x0" ::: "memory");
}

__device__ __forceinline__ float block_reduce_256(float v, float* red) {
    const int tid = threadIdx.x;
    red[tid] = v;
    __syncthreads();
    for (int s = 128; s > 0; s >>= 1) {
        if (tid < s) red[tid] += red[tid + s];
        __syncthreads();
    }
    float r = red[0];
    __syncthreads();
    return r;
}

// ---------------------------------------------------------------------------
// f32 -> split-bf16 (hi + lo) precompute, 8 elements per thread.
// ---------------------------------------------------------------------------
__global__ __launch_bounds__(256)
void convert_kernel(const float* __restrict__ Z, unsigned short* __restrict__ Hi,
                    unsigned short* __restrict__ Lo) {
    const size_t i0 = ((size_t)blockIdx.x * 256 + threadIdx.x) * 8;
    const float4 a = *(const float4*)(Z + i0);
    const float4 b = *(const float4*)(Z + i0 + 4);
    const float x[8] = {a.x, a.y, a.z, a.w, b.x, b.y, b.z, b.w};
    unsigned short h[8], l[8];
#pragma unroll
    for (int e = 0; e < 8; ++e) {
        h[e] = f32_bf16(x[e]);
        l[e] = f32_bf16(x[e] - bf16_f32(h[e]));
    }
    *(uint4*)(Hi + i0) = *(const uint4*)h;
    *(uint4*)(Lo + i0) = *(const uint4*)l;
}

// ---------------------------------------------------------------------------
// Row squared-norms: one block per row.
// ---------------------------------------------------------------------------
__global__ __launch_bounds__(128)
void rownorm_kernel(const float* __restrict__ Z, float* __restrict__ sq) {
    __shared__ float red[128];
    const int row = blockIdx.x;
    const float4 v = ((const float4*)(Z + (size_t)row * kD))[threadIdx.x];
    red[threadIdx.x] = v.x * v.x + v.y * v.y + v.z * v.z + v.w * v.w;
    __syncthreads();
    for (int st = 64; st > 0; st >>= 1) {
        if (threadIdx.x < st) red[threadIdx.x] += red[threadIdx.x + st];
        __syncthreads();
    }
    if (threadIdx.x == 0) sq[row] = red[0];
}

// ---------------------------------------------------------------------------
// Alignment: per-row ||z - z'||, one block per row.
// ---------------------------------------------------------------------------
__global__ __launch_bounds__(128)
void align_kernel(const float* __restrict__ z, const float* __restrict__ zp,
                  float* __restrict__ dout) {
    __shared__ float red[128];
    const int row = blockIdx.x;
    const float4 a = ((const float4*)(z  + (size_t)row * kD))[threadIdx.x];
    const float4 b = ((const float4*)(zp + (size_t)row * kD))[threadIdx.x];
    const float dx = a.x - b.x, dy = a.y - b.y, dz = a.z - b.z, dw = a.w - b.w;
    red[threadIdx.x] = dx * dx + dy * dy + dz * dz + dw * dw;
    __syncthreads();
    for (int st = 64; st > 0; st >>= 1) {
        if (threadIdx.x < st) red[threadIdx.x] += red[threadIdx.x + st];
        __syncthreads();
    }
    if (threadIdx.x == 0) dout[row] = sqrtf(red[0]);
}

// ---------------------------------------------------------------------------
// Shared epilogue: d = sqrt(sq_i + sq_j - 2c), sum exp(-d) over i<j.
// ---------------------------------------------------------------------------
__device__ __forceinline__ float tile_epilogue(const v8f acc[4][2],
                                               const float* __restrict__ sq,
                                               int i0, int j0, int wm, int wn, int lane) {
    const int mo = (lane >> 4) * 8;   // C/D layout: VGPR v -> M = v + mo
    const int nn = lane & 15;         //              lane  -> N = lane & 15
    float lsum = 0.f;
#pragma unroll
    for (int mi = 0; mi < 4; ++mi) {
#pragma unroll
        for (int ni = 0; ni < 2; ++ni) {
            const int gj = j0 + wn * 32 + ni * 16 + nn;
            const float sqj = sq[gj];
#pragma unroll
            for (int v = 0; v < 8; ++v) {
                const int gi = i0 + wm * 64 + mi * 16 + mo + v;
                float d2 = sq[gi] + sqj - 2.0f * acc[mi][ni][v];
                d2 = fmaxf(d2, 0.0f);
                const float e = __expf(-sqrtf(d2));
                lsum += (gi < gj) ? e : 0.0f;
            }
        }
    }
    return lsum;
}

// ---------------------------------------------------------------------------
// MAIN PATH: precomputed split-bf16 + async global->LDS double-buffered GEMM.
// One 128x128 upper-triangle tile of Z Z^T per block; 8 wave32s, each wave a
// 64x32 patch (4x2 WMMA tiles x 3 split-precision WMMAs per K step).
// ---------------------------------------------------------------------------
__global__ __launch_bounds__(256)
void unif_tile_bf16_kernel(const unsigned short* __restrict__ Zhi,
                           const unsigned short* __restrict__ Zlo,
                           const float* __restrict__ sq,
                           float* __restrict__ partial) {
    // [buf][array: 0=Ahi 1=Alo 2=Bhi 3=Blo][tile]
    __shared__ __align__(16) unsigned short lds[2][4][kBT * kST];
    __shared__ float red[256];

    const int tid = threadIdx.x;

    // triangular tile decode (scalar, uniform)
    int rem = blockIdx.x, bi = 0;
    while (rem >= (kNBLK - bi)) { rem -= (kNBLK - bi); ++bi; }
    const int bj = bi + rem;
    const int i0 = bi * kBT, j0 = bj * kBT;

    // loader role: 2 threads per row, each covers 16 contiguous K values (32B)
    const int r  = tid >> 1;
    const int c0 = (tid & 1) * 16;
    const size_t aoff = (size_t)(i0 + r) * kD + c0;   // ushort element offset
    const size_t boff = (size_t)(j0 + r) * kD + c0;
    const unsigned ldsrow = (unsigned)(r * kST + c0) * 2;  // byte offset in a tile

    // compute role
    const int wave = tid >> 5, lane = tid & 31;
    const int wm = wave & 1;            // m base = wm*64
    const int wn = wave >> 1;           // n base = wn*32
    const int frow = lane & 15;
    const int kbA  = (lane >> 4) * 8;   // A-fragment K split (ISA 7.12.2)
    const int kbB  = (lane >> 4) * 8;   // B-fragment K split (ISA 7.12.2)

    v8f acc[4][2];
#pragma unroll
    for (int mi = 0; mi < 4; ++mi)
#pragma unroll
        for (int ni = 0; ni < 2; ++ni)
            acc[mi][ni] = (v8f){0.f, 0.f, 0.f, 0.f, 0.f, 0.f, 0.f, 0.f};

    auto issue_chunk = [&](int buf, int kk) {
        const unsigned base = (unsigned)(uintptr_t)(&lds[buf][0][0]);
        async_copy_b128(base + 0 * kASZ + ldsrow,      Zhi + aoff + kk);
        async_copy_b128(base + 0 * kASZ + ldsrow + 16, Zhi + aoff + kk + 8);
        async_copy_b128(base + 1 * kASZ + ldsrow,      Zlo + aoff + kk);
        async_copy_b128(base + 1 * kASZ + ldsrow + 16, Zlo + aoff + kk + 8);
        async_copy_b128(base + 2 * kASZ + ldsrow,      Zhi + boff + kk);
        async_copy_b128(base + 2 * kASZ + ldsrow + 16, Zhi + boff + kk + 8);
        async_copy_b128(base + 3 * kASZ + ldsrow,      Zlo + boff + kk);
        async_copy_b128(base + 3 * kASZ + ldsrow + 16, Zlo + boff + kk + 8);
    };

    issue_chunk(0, 0);  // prologue

    for (int kc = 0; kc < kNCH; ++kc) {
        wait_async_all();   // this wave's async writes for buf (kc&1) landed
        __syncthreads();    // => all waves' writes landed; prev reads done
        if (kc + 1 < kNCH) issue_chunk((kc + 1) & 1, (kc + 1) * kKC);

        const int cur = kc & 1;
        const unsigned short* Ah = lds[cur][0];
        const unsigned short* Al = lds[cur][1];
        const unsigned short* Bh = lds[cur][2];
        const unsigned short* Bl = lds[cur][3];

        Frag a_hi[4], a_lo[4], b_hi[2], b_lo[2];
#pragma unroll
        for (int mi = 0; mi < 4; ++mi) {
            const int ar = wm * 64 + mi * 16 + frow;
            a_hi[mi].q[0] = *(const uint4*)(&Ah[ar * kST + kbA]);
            a_hi[mi].q[1] = *(const uint4*)(&Ah[ar * kST + 16 + kbA]);
            a_lo[mi].q[0] = *(const uint4*)(&Al[ar * kST + kbA]);
            a_lo[mi].q[1] = *(const uint4*)(&Al[ar * kST + 16 + kbA]);
        }
#pragma unroll
        for (int ni = 0; ni < 2; ++ni) {
            const int br = wn * 32 + ni * 16 + frow;
            b_hi[ni].q[0] = *(const uint4*)(&Bh[br * kST + kbB]);
            b_hi[ni].q[1] = *(const uint4*)(&Bh[br * kST + kbB + 16]);
            b_lo[ni].q[0] = *(const uint4*)(&Bl[br * kST + kbB]);
            b_lo[ni].q[1] = *(const uint4*)(&Bl[br * kST + kbB + 16]);
        }

#pragma unroll
        for (int mi = 0; mi < 4; ++mi)
#pragma unroll
            for (int ni = 0; ni < 2; ++ni) {
                acc[mi][ni] = __builtin_amdgcn_wmma_f32_16x16x32_bf16(
                    false, a_hi[mi].v, false, b_hi[ni].v, (short)0, acc[mi][ni], false, false);
                acc[mi][ni] = __builtin_amdgcn_wmma_f32_16x16x32_bf16(
                    false, a_hi[mi].v, false, b_lo[ni].v, (short)0, acc[mi][ni], false, false);
                acc[mi][ni] = __builtin_amdgcn_wmma_f32_16x16x32_bf16(
                    false, a_lo[mi].v, false, b_hi[ni].v, (short)0, acc[mi][ni], false, false);
            }
    }

    const float lsum = tile_epilogue(acc, sq, i0, j0, wm, wn, lane);
    const float tot = block_reduce_256(lsum, red);
    if (tid == 0) partial[blockIdx.x] = tot;
}

// ---------------------------------------------------------------------------
// FALLBACK PATH (small workspace): convert in-kernel, single-buffered LDS.
// ---------------------------------------------------------------------------
__global__ __launch_bounds__(256)
void unif_tile_f32_kernel(const float* __restrict__ Z, const float* __restrict__ sq,
                          float* __restrict__ partial) {
    __shared__ __align__(16) unsigned short Ahi[kBT * kST];
    __shared__ __align__(16) unsigned short Alo[kBT * kST];
    __shared__ __align__(16) unsigned short Bhi[kBT * kST];
    __shared__ __align__(16) unsigned short Blo[kBT * kST];
    __shared__ float red[256];

    const int tid = threadIdx.x;
    int rem = blockIdx.x, bi = 0;
    while (rem >= (kNBLK - bi)) { rem -= (kNBLK - bi); ++bi; }
    const int bj = bi + rem;
    const int i0 = bi * kBT, j0 = bj * kBT;

    const int r  = tid >> 1;
    const int c0 = (tid & 1) * 16;
    const int wave = tid >> 5, lane = tid & 31;
    const int wm = wave & 1, wn = wave >> 1;
    const int frow = lane & 15;
    const int kbA  = (lane >> 4) * 8;
    const int kbB  = (lane >> 4) * 8;

    v8f acc[4][2];
#pragma unroll
    for (int mi = 0; mi < 4; ++mi)
#pragma unroll
        for (int ni = 0; ni < 2; ++ni)
            acc[mi][ni] = (v8f){0.f, 0.f, 0.f, 0.f, 0.f, 0.f, 0.f, 0.f};

    const float* Arow = Z + (size_t)(i0 + r) * kD + c0;
    const float* Brow = Z + (size_t)(j0 + r) * kD + c0;

    for (int kk = 0; kk < kD; kk += kKC) {
        float4 av[4], bv[4];
#pragma unroll
        for (int q = 0; q < 4; ++q) {
            av[q] = *(const float4*)(Arow + kk + q * 4);
            bv[q] = *(const float4*)(Brow + kk + q * 4);
        }
        if (kk + kKC < kD) {
            __builtin_prefetch(Arow + kk + kKC, 0, 3);
            __builtin_prefetch(Brow + kk + kKC, 0, 3);
        }
        unsigned short ah[16], al[16], bh[16], bl[16];
#pragma unroll
        for (int q = 0; q < 4; ++q) {
            const float aa[4] = {av[q].x, av[q].y, av[q].z, av[q].w};
            const float bb[4] = {bv[q].x, bv[q].y, bv[q].z, bv[q].w};
#pragma unroll
            for (int e = 0; e < 4; ++e) {
                const int idx = q * 4 + e;
                unsigned short h = f32_bf16(aa[e]);
                ah[idx] = h;
                al[idx] = f32_bf16(aa[e] - bf16_f32(h));
                h = f32_bf16(bb[e]);
                bh[idx] = h;
                bl[idx] = f32_bf16(bb[e] - bf16_f32(h));
            }
        }

        __syncthreads();
        *(uint4*)(&Ahi[r * kST + c0])     = *(const uint4*)(ah + 0);
        *(uint4*)(&Ahi[r * kST + c0 + 8]) = *(const uint4*)(ah + 8);
        *(uint4*)(&Alo[r * kST + c0])     = *(const uint4*)(al + 0);
        *(uint4*)(&Alo[r * kST + c0 + 8]) = *(const uint4*)(al + 8);
        *(uint4*)(&Bhi[r * kST + c0])     = *(const uint4*)(bh + 0);
        *(uint4*)(&Bhi[r * kST + c0 + 8]) = *(const uint4*)(bh + 8);
        *(uint4*)(&Blo[r * kST + c0])     = *(const uint4*)(bl + 0);
        *(uint4*)(&Blo[r * kST + c0 + 8]) = *(const uint4*)(bl + 8);
        __syncthreads();

        Frag a_hi[4], a_lo[4], b_hi[2], b_lo[2];
#pragma unroll
        for (int mi = 0; mi < 4; ++mi) {
            const int ar = wm * 64 + mi * 16 + frow;
            a_hi[mi].q[0] = *(const uint4*)(&Ahi[ar * kST + kbA]);
            a_hi[mi].q[1] = *(const uint4*)(&Ahi[ar * kST + 16 + kbA]);
            a_lo[mi].q[0] = *(const uint4*)(&Alo[ar * kST + kbA]);
            a_lo[mi].q[1] = *(const uint4*)(&Alo[ar * kST + 16 + kbA]);
        }
#pragma unroll
        for (int ni = 0; ni < 2; ++ni) {
            const int br = wn * 32 + ni * 16 + frow;
            b_hi[ni].q[0] = *(const uint4*)(&Bhi[br * kST + kbB]);
            b_hi[ni].q[1] = *(const uint4*)(&Bhi[br * kST + kbB + 16]);
            b_lo[ni].q[0] = *(const uint4*)(&Blo[br * kST + kbB]);
            b_lo[ni].q[1] = *(const uint4*)(&Blo[br * kST + kbB + 16]);
        }

#pragma unroll
        for (int mi = 0; mi < 4; ++mi)
#pragma unroll
            for (int ni = 0; ni < 2; ++ni) {
                acc[mi][ni] = __builtin_amdgcn_wmma_f32_16x16x32_bf16(
                    false, a_hi[mi].v, false, b_hi[ni].v, (short)0, acc[mi][ni], false, false);
                acc[mi][ni] = __builtin_amdgcn_wmma_f32_16x16x32_bf16(
                    false, a_hi[mi].v, false, b_lo[ni].v, (short)0, acc[mi][ni], false, false);
                acc[mi][ni] = __builtin_amdgcn_wmma_f32_16x16x32_bf16(
                    false, a_lo[mi].v, false, b_hi[ni].v, (short)0, acc[mi][ni], false, false);
            }
    }

    const float lsum = tile_epilogue(acc, sq, i0, j0, wm, wn, lane);
    const float tot = block_reduce_256(lsum, red);
    if (tid == 0) partial[blockIdx.x] = tot;
}

// ---------------------------------------------------------------------------
// Final combine: align mean + 0.5*(log-mean-exp terms)
// ---------------------------------------------------------------------------
__global__ __launch_bounds__(256)
void finalize_kernel(const float* __restrict__ dAlign,
                     const float* __restrict__ p0,
                     const float* __restrict__ p1,
                     float* __restrict__ out) {
    __shared__ float red[256];
    const int tid = threadIdx.x;

    float sA = 0.f;
    for (int i = tid; i < kN; i += 256) sA += dAlign[i];
    sA = block_reduce_256(sA, red);

    float s0 = 0.f;
    for (int i = tid; i < kNTRI; i += 256) s0 += p0[i];
    s0 = block_reduce_256(s0, red);

    float s1 = 0.f;
    for (int i = tid; i < kNTRI; i += 256) s1 += p1[i];
    s1 = block_reduce_256(s1, red);

    if (tid == 0) {
        const float npairs = (float)kN * (float)(kN - 1) * 0.5f;  // 33550336
        out[0] = sA / (float)kN + 0.5f * (logf(s0 / npairs) + logf(s1 / npairs));
    }
}

extern "C" void kernel_launch(void* const* d_in, const int* in_sizes, int n_in,
                              void* d_out, int out_size, void* d_ws, size_t ws_size,
                              hipStream_t stream) {
    (void)in_sizes; (void)n_in; (void)out_size;
    const float* z  = (const float*)d_in[0];
    const float* zp = (const float*)d_in[1];
    float* out = (float*)d_out;

    const size_t nElem    = (size_t)kN * kD;                       // 4M
    const size_t bfBytes  = 4 * nElem * sizeof(unsigned short);    // 33.5 MB
    const size_t smallFl  = (size_t)3 * kN + 2 * kNTRI;            // floats

    if (ws_size >= bfBytes + smallFl * sizeof(float)) {
        // ---- async-TDM-style path with precomputed split-bf16 ----
        unsigned short* zhi0 = (unsigned short*)d_ws;
        unsigned short* zlo0 = zhi0 + nElem;
        unsigned short* zhi1 = zlo0 + nElem;
        unsigned short* zlo1 = zhi1 + nElem;
        float* fs  = (float*)((char*)d_ws + bfBytes);
        float* sq0 = fs;
        float* sq1 = fs + kN;
        float* dAl = fs + 2 * kN;
        float* p0  = fs + 3 * kN;
        float* p1  = p0 + kNTRI;

        const int cvtBlocks = (int)(nElem / (256 * 8));  // 2048
        convert_kernel<<<cvtBlocks, 256, 0, stream>>>(z,  zhi0, zlo0);
        convert_kernel<<<cvtBlocks, 256, 0, stream>>>(zp, zhi1, zlo1);
        rownorm_kernel<<<kN, 128, 0, stream>>>(z,  sq0);
        rownorm_kernel<<<kN, 128, 0, stream>>>(zp, sq1);
        align_kernel<<<kN, 128, 0, stream>>>(z, zp, dAl);
        unif_tile_bf16_kernel<<<kNTRI, 256, 0, stream>>>(zhi0, zlo0, sq0, p0);
        unif_tile_bf16_kernel<<<kNTRI, 256, 0, stream>>>(zhi1, zlo1, sq1, p1);
        finalize_kernel<<<1, 256, 0, stream>>>(dAl, p0, p1, out);
    } else {
        // ---- small-workspace fallback ----
        float* ws  = (float*)d_ws;
        float* sq0 = ws;
        float* sq1 = ws + kN;
        float* dAl = ws + 2 * kN;
        float* p0  = ws + 3 * kN;
        float* p1  = p0 + kNTRI;

        rownorm_kernel<<<kN, 128, 0, stream>>>(z,  sq0);
        rownorm_kernel<<<kN, 128, 0, stream>>>(zp, sq1);
        align_kernel<<<kN, 128, 0, stream>>>(z, zp, dAl);
        unif_tile_f32_kernel<<<kNTRI, 256, 0, stream>>>(z,  sq0, p0);
        unif_tile_f32_kernel<<<kNTRI, 256, 0, stream>>>(zp, sq1, p1);
        finalize_kernel<<<1, 256, 0, stream>>>(dAl, p0, p1, out);
    }
}
